// LSTMCRFModel_25262997635532
// MI455X (gfx1250) — matrix operand order repaired and checked by reading, not compiled, not run on hardware
//
#include <hip/hip_runtime.h>
#include <hip/hip_bf16.h>
#include <math.h>
#include <stdint.h>

typedef __attribute__((ext_vector_type(16))) _Float16 v16h;
typedef __attribute__((ext_vector_type(8)))  float    v8f;

#define BB    32
#define SS    256
#define EMBD  300
#define HIDD  300
#define G4    1200      // 4*HID gate width
#define KP    320       // K padded to multiple of 32
#define NTAGS 9
#define MROWS (SS*BB)   // 8192

union Frag16 { uint4 q[2]; v16h v; };

// A fragment (16x32 f16, MxK): lane -> m = lane&15, half = lane>>4.
// element j: K = 16*(j>>3) + 8*half + (j&7)  => two contiguous 16B chunks.
__device__ __forceinline__ v16h load_a_frag(const _Float16* A, int lda, int m, int kb, int half) {
  Frag16 f;
  const _Float16* p = A + m * lda + kb + 8 * half;
  f.q[0] = *(const uint4*)(p);
  f.q[1] = *(const uint4*)(p + 16);
  return f.v;
}

// B fragment (32x16 f16, KxN) with B[k][n] = W[n][k] (row-major W, stride ldb):
// lane -> n = lane&15, half = lane>>4; element j: K = kb + 16*(j>>3) + 8*half + (j&7).
__device__ __forceinline__ v16h load_b_frag(const _Float16* W, int ldb, int n, int kb, int half) {
  Frag16 f;
  const _Float16* p = W + (size_t)n * ldb + kb + 8 * half;
  f.q[0] = *(const uint4*)(p);
  f.q[1] = *(const uint4*)(p + 16);
  return f.v;
}

__device__ __forceinline__ float sigm(float x) { return 1.0f / (1.0f + expf(-x)); }

// Async copy of one 16B chunk (per lane) from global to LDS; tracked by ASYNCcnt.
__device__ __forceinline__ void async_g2lds_b128(unsigned lds_off, const void* gptr) {
  asm volatile("global_load_async_to_lds_b128 %0, %1, off"
               :
               : "v"(lds_off), "v"((unsigned long long)(uintptr_t)gptr)
               : "memory");
}
__device__ __forceinline__ void wait_asynccnt0() {
#if __has_builtin(__builtin_amdgcn_s_wait_asynccnt)
  __builtin_amdgcn_s_wait_asynccnt(0);
#else
  asm volatile("s_wait_asynccnt 0x0" ::: "memory");
#endif
}

// ---------------------------------------------------------------------------
// 1) Convert the four LSTM weight matrices (1200x300 f32) to f16, K-padded to 320.
//    Layout in w16: [0]=w_ih_f, [1]=w_hh_f, [2]=w_ih_b, [3]=w_hh_b
// ---------------------------------------------------------------------------
__global__ void k_convert_weights(const float* __restrict__ w0, const float* __restrict__ w1,
                                  const float* __restrict__ w2, const float* __restrict__ w3,
                                  _Float16* __restrict__ w16) {
  int idx = blockIdx.x * blockDim.x + threadIdx.x;
  const int per = G4 * KP;
  if (idx >= 4 * per) return;
  int a = idx / per, r = idx % per;
  int row = r / KP, k = r % KP;
  const float* src = (a == 0) ? w0 : (a == 1) ? w1 : (a == 2) ? w2 : w3;
  float v = (k < HIDD) ? src[row * HIDD + k] : 0.0f;  // EMB == HID == 300
  w16[idx] = (_Float16)v;
}

// ---------------------------------------------------------------------------
// 2) Embedding gather -> xs16 in (S,B,KP) f16 layout, zero-padded K 300..319
// ---------------------------------------------------------------------------
__global__ void k_embed(const int* __restrict__ x, const float* __restrict__ emb,
                        _Float16* __restrict__ xs16) {
  int idx = blockIdx.x * blockDim.x + threadIdx.x;
  if (idx >= MROWS * KP) return;
  int row = idx / KP, k = idx % KP;
  int s = row / BB, b = row % BB;
  int tok = x[b * SS + s];
  float v = (k < EMBD) ? emb[(size_t)tok * EMBD + k] : 0.0f;
  xs16[idx] = (_Float16)v;
}

// ---------------------------------------------------------------------------
// 3) Gate precompute: gx[dir] = xs @ w_ih^T + b_ih.  M=8192, N=1200, K=320.
//    Block = 5 waves; block tile = 64(M) x 80(N); A async-staged in LDS;
//    each wave: 1 N-subtile, 4 M-subtiles -> B-fragment reused 4x.
// ---------------------------------------------------------------------------
__global__ void __launch_bounds__(160)
k_gemm_gates(const _Float16* __restrict__ xs16, const _Float16* __restrict__ w16,
             const float* __restrict__ b_ih_f, const float* __restrict__ b_ih_b,
             float* __restrict__ gx) {
  __shared__ __align__(16) _Float16 As[64 * KP];  // 40 KB
  const int dir  = blockIdx.z;
  const int wave = threadIdx.x >> 5, lane = threadIdx.x & 31;
  const int m0 = blockIdx.x * 64;
  const int n0 = blockIdx.y * 80 + wave * 16;
  const _Float16* W    = w16 + (size_t)(dir * 2) * (G4 * KP);   // w_ih for this dir
  const float*    bias = dir ? b_ih_b : b_ih_f;

  // async stage A tile: 64 rows x 320 halfs = 2560 uint4 (16 per thread)
  {
    const uint4* srcA = (const uint4*)(xs16 + (size_t)m0 * KP);
    const unsigned lbase = (unsigned)(uintptr_t)(&As[0]);
#pragma unroll
    for (int k = 0; k < 16; ++k) {
      const int i = threadIdx.x + 160 * k;
      async_g2lds_b128(lbase + (unsigned)i * 16u, srcA + i);
    }
    wait_asynccnt0();
  }
  __syncthreads();

  const int half = lane >> 4, l15 = lane & 15;
  const int n = n0 + l15;
  v8f acc0 = {}, acc1 = {}, acc2 = {}, acc3 = {};
#pragma unroll
  for (int kb = 0; kb < KP; kb += 32) {
    v16h b  = load_b_frag(W, KP, n, kb, half);
    v16h a0 = load_a_frag(As, KP,  0 + l15, kb, half);
    v16h a1 = load_a_frag(As, KP, 16 + l15, kb, half);
    v16h a2 = load_a_frag(As, KP, 32 + l15, kb, half);
    v16h a3 = load_a_frag(As, KP, 48 + l15, kb, half);
    acc0 = __builtin_amdgcn_wmma_f32_16x16x32_f16(false, a0, false, b, (short)0, acc0, false, false);
    acc1 = __builtin_amdgcn_wmma_f32_16x16x32_f16(false, a1, false, b, (short)0, acc1, false, false);
    acc2 = __builtin_amdgcn_wmma_f32_16x16x32_f16(false, a2, false, b, (short)0, acc2, false, false);
    acc3 = __builtin_amdgcn_wmma_f32_16x16x32_f16(false, a3, false, b, (short)0, acc3, false, false);
  }
  const float bn = bias[n];
  float* gout = gx + (size_t)dir * MROWS * G4;
#pragma unroll
  for (int r = 0; r < 8; ++r) {
    const int mr = 8 * half + r;
    gout[(size_t)(m0 +  0 + mr) * G4 + n] = acc0[r] + bn;
    gout[(size_t)(m0 + 16 + mr) * G4 + n] = acc1[r] + bn;
    gout[(size_t)(m0 + 32 + mr) * G4 + n] = acc2[r] + bn;
    gout[(size_t)(m0 + 48 + mr) * G4 + n] = acc3[r] + bn;
  }
}

// ---------------------------------------------------------------------------
// 4) Recurrent BiLSTM: one workgroup per direction (32 waves), 256 steps.
//    gates kept in LDS (f16), c-state in registers, h in LDS (f16, KP-padded).
//    Per step: 75 N-tiles over 32 waves, 2 M-tiles per B-fragment.
// ---------------------------------------------------------------------------
__global__ void __launch_bounds__(1024)
k_lstm(const _Float16* __restrict__ w16, const float* __restrict__ gx,
       const float* __restrict__ b_hh_f, const float* __restrict__ b_hh_b,
       float* __restrict__ hcat) {
  __shared__ __align__(16) _Float16 h16[BB * KP];      // 20 KB
  __shared__ __align__(16) _Float16 gates16[BB * G4];  // 76.8 KB
  const int dir = blockIdx.x;
  const int tid = threadIdx.x;
  const _Float16* W   = w16 + (size_t)(dir * 2 + 1) * (G4 * KP);  // w_hh for this dir
  const float*    gxd = gx + (size_t)dir * MROWS * G4;
  const float*    bhh = dir ? b_hh_b : b_hh_f;

  for (int i = tid; i < BB * KP; i += 1024) h16[i] = (_Float16)0.0f;
  float creg[10];
#pragma unroll
  for (int k = 0; k < 10; ++k) creg[k] = 0.0f;  // c state, element u = tid + k*1024
  __syncthreads();

  const int wave = tid >> 5, lane = tid & 31;
  const int half = lane >> 4, l15 = lane & 15;

  for (int it = 0; it < SS; ++it) {
    const int t = dir ? (SS - 1 - it) : it;
    // ---- phase 1: gates = h16 @ w_hh^T + gx[t] + b_hh  (M=32,N=1200,K=320)
    for (int nt = wave; nt < 75; nt += 32) {
      const int n = nt * 16 + l15;
      v8f acc0 = {}, acc1 = {};
#pragma unroll
      for (int kb = 0; kb < KP; kb += 32) {
        v16h b  = load_b_frag(W, KP, n, kb, half);
        v16h a0 = load_a_frag(h16, KP,  0 + l15, kb, half);
        v16h a1 = load_a_frag(h16, KP, 16 + l15, kb, half);
        acc0 = __builtin_amdgcn_wmma_f32_16x16x32_f16(false, a0, false, b, (short)0, acc0, false, false);
        acc1 = __builtin_amdgcn_wmma_f32_16x16x32_f16(false, a1, false, b, (short)0, acc1, false, false);
      }
      const float bn = bhh[n];
#pragma unroll
      for (int r = 0; r < 8; ++r) {
        const int r0 = 8 * half + r;
        const int r1 = 16 + r0;
        gates16[r0 * G4 + n] = (_Float16)(acc0[r] + bn + gxd[(size_t)(t * BB + r0) * G4 + n]);
        gates16[r1 * G4 + n] = (_Float16)(acc1[r] + bn + gxd[(size_t)(t * BB + r1) * G4 + n]);
      }
    }
    __syncthreads();
    // ---- phase 2: elementwise LSTM cell update (c in registers)
#pragma unroll
    for (int k = 0; k < 10; ++k) {
      const int u = tid + k * 1024;
      if (u < BB * HIDD) {
        const int b = u / HIDD, j = u % HIDD;
        const float ig = (float)gates16[b * G4 + j];
        const float fg = (float)gates16[b * G4 + 300 + j];
        const float gg = (float)gates16[b * G4 + 600 + j];
        const float og = (float)gates16[b * G4 + 900 + j];
        const float cn = sigm(fg) * creg[k] + sigm(ig) * tanhf(gg);
        const float hn = sigm(og) * tanhf(cn);
        creg[k] = cn;
        h16[b * KP + j] = (_Float16)hn;
        hcat[(size_t)(b * SS + t) * (2 * HIDD) + dir * HIDD + j] = hn;
      }
    }
    __syncthreads();
  }
}

// ---------------------------------------------------------------------------
// 5) Linear (600 -> 9) + softmax -> probs (d_out); weights staged in LDS
// ---------------------------------------------------------------------------
__global__ void __launch_bounds__(256)
k_linear_softmax(const float* __restrict__ hcat, const float* __restrict__ lin_w,
                 const float* __restrict__ lin_b, float* __restrict__ probs) {
  __shared__ float wsm[NTAGS * 2 * HIDD];  // 21.6 KB
  __shared__ float bsm[NTAGS];
  for (int i = threadIdx.x; i < NTAGS * 2 * HIDD; i += 256) wsm[i] = lin_w[i];
  if (threadIdx.x < NTAGS) bsm[threadIdx.x] = lin_b[threadIdx.x];
  __syncthreads();

  int row = blockIdx.x * blockDim.x + threadIdx.x;  // row = b*S + s
  if (row >= BB * SS) return;
  const float* hr = hcat + (size_t)row * (2 * HIDD);
  float acc[NTAGS];
#pragma unroll
  for (int j = 0; j < NTAGS; ++j) acc[j] = bsm[j];
  for (int k = 0; k < 2 * HIDD; ++k) {
    const float hv = hr[k];
#pragma unroll
    for (int j = 0; j < NTAGS; ++j) acc[j] += hv * wsm[j * (2 * HIDD) + k];
  }
  float m = acc[0];
#pragma unroll
  for (int j = 1; j < NTAGS; ++j) m = fmaxf(m, acc[j]);
  float s = 0.0f;
#pragma unroll
  for (int j = 0; j < NTAGS; ++j) { acc[j] = expf(acc[j] - m); s += acc[j]; }
  const float inv = 1.0f / s;
#pragma unroll
  for (int j = 0; j < NTAGS; ++j) probs[(size_t)row * NTAGS + j] = acc[j] * inv;
}

// ---------------------------------------------------------------------------
// 6) CRF gold score + forward algorithm -> scalar loss (wave32 reduction)
// ---------------------------------------------------------------------------
__global__ void k_crf(const float* __restrict__ probs, const int* __restrict__ y,
                      const float* __restrict__ start_t, const float* __restrict__ end_t,
                      const float* __restrict__ trans, float* __restrict__ loss_out) {
  const int b = threadIdx.x;
  float local = 0.0f;
  if (b < BB) {
    const int* yb = y + b * SS;
    const float* em = probs + (size_t)b * SS * NTAGS;
    // gold path score
    const bool m0 = (yb[0] != -1);
    int tprev = m0 ? yb[0] : 0;
    float score = start_t[tprev] + em[tprev];
    int cnt = m0 ? 1 : 0;
    for (int s = 1; s < SS; ++s) {
      const bool ms = (yb[s] != -1);
      const int tg = ms ? yb[s] : 0;
      if (ms) { score += trans[tprev * NTAGS + tg] + em[s * NTAGS + tg]; ++cnt; }
      tprev = tg;
    }
    int last_idx = cnt - 1; if (last_idx < 0) last_idx = 0;
    const int lt = (yb[last_idx] != -1) ? yb[last_idx] : 0;
    score += end_t[lt];
    // forward algorithm (log partition)
    float alpha[NTAGS];
#pragma unroll
    for (int j = 0; j < NTAGS; ++j) alpha[j] = start_t[j] + em[j];
    for (int s = 1; s < SS; ++s) {
      if (yb[s] != -1) {
        float na[NTAGS];
        for (int j = 0; j < NTAGS; ++j) {
          float mx = alpha[0] + trans[j];
          for (int i = 1; i < NTAGS; ++i) mx = fmaxf(mx, alpha[i] + trans[i * NTAGS + j]);
          float ss = 0.0f;
          for (int i = 0; i < NTAGS; ++i) ss += expf(alpha[i] + trans[i * NTAGS + j] - mx);
          na[j] = mx + logf(ss) + em[s * NTAGS + j];
        }
#pragma unroll
        for (int j = 0; j < NTAGS; ++j) alpha[j] = na[j];
      }
    }
    float mx = alpha[0] + end_t[0];
#pragma unroll
    for (int j = 1; j < NTAGS; ++j) mx = fmaxf(mx, alpha[j] + end_t[j]);
    float ss = 0.0f;
#pragma unroll
    for (int j = 0; j < NTAGS; ++j) ss += expf(alpha[j] + end_t[j] - mx);
    local = score - (mx + logf(ss));
  }
  for (int off = 16; off > 0; off >>= 1) local += __shfl_xor(local, off, 32);
  if (threadIdx.x == 0) loss_out[0] = -local;
}

// ---------------------------------------------------------------------------
extern "C" void kernel_launch(void* const* d_in, const int* in_sizes, int n_in,
                              void* d_out, int out_size, void* d_ws, size_t ws_size,
                              hipStream_t stream) {
  (void)in_sizes; (void)n_in; (void)out_size; (void)ws_size;
  const int*   x       = (const int*)  d_in[0];
  const int*   y       = (const int*)  d_in[1];
  const float* emb     = (const float*)d_in[2];
  const float* w_ih_f  = (const float*)d_in[3];
  const float* w_hh_f  = (const float*)d_in[4];
  const float* b_ih_f  = (const float*)d_in[5];
  const float* b_hh_f  = (const float*)d_in[6];
  const float* w_ih_b  = (const float*)d_in[7];
  const float* w_hh_b  = (const float*)d_in[8];
  const float* b_ih_b  = (const float*)d_in[9];
  const float* b_hh_b  = (const float*)d_in[10];
  const float* lin_w   = (const float*)d_in[11];
  const float* lin_b   = (const float*)d_in[12];
  const float* start_t = (const float*)d_in[13];
  const float* end_t   = (const float*)d_in[14];
  const float* trans   = (const float*)d_in[15];

  float* probs = (float*)d_out;              // 32*256*9
  float* loss  = probs + BB * SS * NTAGS;    // +1 scalar

  // workspace partition (256B-aligned chunks)
  char* ws = (char*)d_ws;
  size_t off = 0;
  auto wsalloc = [&](size_t bytes) -> void* {
    void* p = ws + off;
    off += (bytes + 255) & ~(size_t)255;
    return p;
  };
  _Float16* xs16 = (_Float16*)wsalloc((size_t)MROWS * KP * sizeof(_Float16));      // 5.2 MB
  _Float16* w16  = (_Float16*)wsalloc((size_t)4 * G4 * KP * sizeof(_Float16));     // 3.1 MB
  float*    gx   = (float*)   wsalloc((size_t)2 * MROWS * G4 * sizeof(float));     // 78.6 MB
  float*    hcat = (float*)   wsalloc((size_t)BB * SS * 2 * HIDD * sizeof(float)); // 19.7 MB

  k_convert_weights<<<dim3((4 * G4 * KP + 255) / 256), dim3(256), 0, stream>>>(
      w_ih_f, w_hh_f, w_ih_b, w_hh_b, w16);
  k_embed<<<dim3((MROWS * KP + 255) / 256), dim3(256), 0, stream>>>(x, emb, xs16);
  k_gemm_gates<<<dim3(MROWS / 64, 15, 2), dim3(160), 0, stream>>>(
      xs16, w16, b_ih_f, b_ih_b, gx);
  k_lstm<<<dim3(2), dim3(1024), 0, stream>>>(w16, gx, b_hh_f, b_hh_b, hcat);
  k_linear_softmax<<<dim3((BB * SS + 255) / 256), dim3(256), 0, stream>>>(
      hcat, lin_w, lin_b, probs);
  k_crf<<<dim3(1), dim3(32), 0, stream>>>(probs, y, start_t, end_t, trans, loss);
}
